// SambaForSummarization_7421703488077
// MI455X (gfx1250) — matrix operand order repaired and checked by reading, not compiled, not run on hardware
//
#include <hip/hip_runtime.h>
#include <math.h>

typedef __attribute__((ext_vector_type(16))) __bf16 v16bf;
typedef __attribute__((ext_vector_type(8)))  float  v8f;

#define VOCAB   30522
#define DMODEL  512
#define DINNER  1024
#define HDIM    64
#define NHEADS  8
#define WINDOW_ 512

// ---------------------------------------------------------------------------
// Kernels
// ---------------------------------------------------------------------------

// Transpose + convert fp32 weight [K][N] -> bf16 [Npad][K] (zero-padded cols).
__global__ void wconv_kernel(const float* __restrict__ W, __bf16* __restrict__ Wt,
                             int K, int N, int Npad) {
  long i = (long)blockIdx.x * blockDim.x + threadIdx.x;
  long tot = (long)Npad * K;
  if (i >= tot) return;
  int n = (int)(i / K);
  int k = (int)(i - (long)n * K);
  float v = (n < N) ? W[(size_t)k * N + n] : 0.f;
  Wt[i] = (__bf16)v;
}

// C[M][N] = act(A[M][K] x W + bias) (+ res). One wave per 16x64 tile:
// one A fragment (fp32 -> bf16 in-register) amortized over 4 B fragments /
// 4 v_wmma_f32_16x16x32_bf16 per K-step. W pre-converted bf16 [Npad][K].
__global__ __launch_bounds__(256)
void gemm_bf16_kernel(const float* __restrict__ A, const __bf16* __restrict__ Wt,
                      const float* __restrict__ bias, const float* __restrict__ res,
                      float* __restrict__ C, int M, int K, int N, int Npad, int act) {
  int wave = threadIdx.x >> 5;
  int lane = threadIdx.x & 31;
  int tilesN = Npad >> 6;                 // 64-wide N tiles
  int totalTiles = (M >> 4) * tilesN;
  int tile = blockIdx.x * 8 + wave;
  if (tile >= totalTiles) return;
  int tm = tile / tilesN;
  int tn = tile - tm * tilesN;
  int row0 = tm << 4;
  int col0 = tn << 6;
  int mn = lane & 15;   // A row within tile == B col within 16-wide subtile
  int hi = lane >> 4;   // lane-group select (K halves)
  const float*  Ar = A  + (size_t)(row0 + mn) * K;
  const __bf16* W0 = Wt + (size_t)(col0 + mn) * K;
  int kb8  = hi << 3;   // unused; A also uses kb16
  int kb16 = hi << 4;   // B: 0 or 16
  v8f acc[4] = {{}, {}, {}, {}};
  for (int k0 = 0; k0 < K; k0 += 32) {
    // prefetch next K block (emits global_prefetch_b8)
    if (k0 + 32 < K) {
      __builtin_prefetch(Ar + k0 + 32, 0, 1);
      __builtin_prefetch(W0 + k0 + 32, 0, 1);
    }
    v16bf a;
#pragma unroll
    for (int i = 0; i < 8; ++i) {
      a[i]     = (__bf16)Ar[k0 + kb16 + i];
      a[i + 8] = (__bf16)Ar[k0 + kb16 + 8 + i];
    }
    v16bf b0, b1, b2, b3;
#pragma unroll
    for (int i = 0; i < 16; ++i) {
      b0[i] = W0[k0 + kb16 + i];
      b1[i] = W0[(size_t)16 * K + k0 + kb16 + i];
      b2[i] = W0[(size_t)32 * K + k0 + kb16 + i];
      b3[i] = W0[(size_t)48 * K + k0 + kb16 + i];
    }
    acc[0] = __builtin_amdgcn_wmma_f32_16x16x32_bf16(false, a, false, b0, (short)0, acc[0], false, false);
    acc[1] = __builtin_amdgcn_wmma_f32_16x16x32_bf16(false, a, false, b1, (short)0, acc[1], false, false);
    acc[2] = __builtin_amdgcn_wmma_f32_16x16x32_bf16(false, a, false, b2, (short)0, acc[2], false, false);
    acc[3] = __builtin_amdgcn_wmma_f32_16x16x32_bf16(false, a, false, b3, (short)0, acc[3], false, false);
  }
#pragma unroll
  for (int j = 0; j < 4; ++j) {
    int n = col0 + (j << 4) + mn;
    if (n >= N) continue;
    float bv = bias ? bias[n] : 0.f;
#pragma unroll
    for (int r = 0; r < 8; ++r) {
      int row = row0 + (hi << 3) + r;
      float v = acc[j][r] + bv;
      if (act == 1)      v = v / (1.f + __expf(-v));                       // silu
      else if (act == 2) v = fmaxf(v, 0.f) + log1pf(__expf(-fabsf(v)));    // softplus
      if (res) v += res[(size_t)row * N + n];
      C[(size_t)row * N + n] = v;
    }
  }
}

// LayerNorm over D=512, one wave per token.
__global__ void ln_kernel(const float* __restrict__ x, const float* __restrict__ s,
                          const float* __restrict__ b, float* __restrict__ o, int M) {
  int token = blockIdx.x * 8 + (threadIdx.x >> 5);
  int lane = threadIdx.x & 31;
  if (token >= M) return;
  const float* xr = x + (size_t)token * DMODEL;
  float vals[16];
  float sum = 0.f;
#pragma unroll
  for (int i = 0; i < 16; ++i) { vals[i] = xr[lane * 16 + i]; sum += vals[i]; }
  for (int off = 16; off >= 1; off >>= 1) sum += __shfl_xor(sum, off, 32);
  float mu = sum * (1.f / DMODEL);
  float vs = 0.f;
#pragma unroll
  for (int i = 0; i < 16; ++i) { float d = vals[i] - mu; vs += d * d; }
  for (int off = 16; off >= 1; off >>= 1) vs += __shfl_xor(vs, off, 32);
  float rinv = rsqrtf(vs * (1.f / DMODEL) + 1e-6f);
  float* orow = o + (size_t)token * DMODEL;
#pragma unroll
  for (int i = 0; i < 16; ++i) {
    int c = lane * 16 + i;
    orow[c] = (vals[i] - mu) * rinv * s[c] + b[c];
  }
}

// Embedding gather: out[tok][c] = emb[ids[tok]][c]
__global__ void embed_kernel(const float* __restrict__ emb, const int* __restrict__ ids,
                             float* __restrict__ out, int Mtok) {
  long i = (long)blockIdx.x * blockDim.x + threadIdx.x;
  if (i >= (long)Mtok * DMODEL) return;
  int row = (int)(i >> 9);
  int c = (int)(i & 511);
  out[i] = emb[(size_t)ids[row] * DMODEL + c];
}

// im2col for D_CONV=4, 'SAME' (taps t-1..t+2), input = x_z (cols 0..1023 of XZ).
__global__ void im2col_kernel(const float* __restrict__ xz, float* __restrict__ X2,
                              int Bb, int L) {
  long i = (long)blockIdx.x * blockDim.x + threadIdx.x;
  if (i >= (long)Bb * L * 4096) return;
  int c = (int)(i & 4095);
  int token = (int)(i >> 12);
  int w = c >> 10;
  int ci = c & 1023;
  int b = token / L;
  int t = token - b * L;
  int tt = t + w - 1;
  float v = 0.f;
  if (tt >= 0 && tt < L) v = xz[((size_t)(b * L + tt)) * 2048 + ci];
  X2[i] = v;
}

// Mamba selective scan: per (b, n) channel, 16 states, decay 0.95^(t+1).
__global__ void scan_kernel(const float* __restrict__ delta, const float* __restrict__ xc,
                            const float* __restrict__ Bm, const float* __restrict__ Cm,
                            float* __restrict__ y, int Bb, int L) {
  int idx = blockIdx.x * blockDim.x + threadIdx.x;
  if (idx >= Bb * DINNER) return;
  int b = idx >> 10;
  int n = idx & 1023;
  float st[16];
#pragma unroll
  for (int s = 0; s < 16; ++s) st[s] = 0.f;
  float decay = 0.95f;
  for (int t = 0; t < L; ++t) {
    size_t tok = (size_t)(b * L + t);
    float coef = delta[tok * DINNER + n] * xc[tok * DINNER + n] * decay;
    const float* Bt = Bm + tok * 16;
    const float* Ct = Cm + tok * 16;
    float acc = 0.f;
#pragma unroll
    for (int s = 0; s < 16; ++s) { st[s] += coef * Bt[s]; acc += st[s] * Ct[s]; }
    y[tok * DINNER + n] = acc;
    decay *= 0.95f;
  }
}

// y *= silu(z), z = XZ[:, 1024:2048]
__global__ void gate_kernel(float* __restrict__ y, const float* __restrict__ xz, long n) {
  long i = (long)blockIdx.x * blockDim.x + threadIdx.x;
  if (i >= n) return;
  int row = (int)(i >> 10);
  int c = (int)(i & 1023);
  float z = xz[(size_t)row * 2048 + 1024 + c];
  y[i] *= z / (1.f + __expf(-z));
}

// Windowed softmax attention. Grid (Lq, H, B), 64 threads/block (one query).
// window < 0 -> full attention. Q/K/V are [B*L, 512] with head h at cols h*64.
__global__ void attn_kernel(const float* __restrict__ Q, const float* __restrict__ Kb,
                            const float* __restrict__ Vb, float* __restrict__ O,
                            int Lq, int Lk, int window) {
  __shared__ float s_q[HDIM];
  __shared__ float s_sc[2048];
  __shared__ float s_red[64];
  int q = blockIdx.x, h = blockIdx.y, b = blockIdx.z;
  int tid = threadIdx.x;
  const float* qp = Q + ((size_t)(b * Lq + q) * DMODEL) + h * HDIM;
  s_q[tid] = qp[tid];
  __syncthreads();
  int j0 = 0, j1 = Lk - 1;
  if (window >= 0) {
    j0 = q - window; if (j0 < 0) j0 = 0;
    j1 = q + window; if (j1 > Lk - 1) j1 = Lk - 1;
  }
  int cnt = j1 - j0 + 1;
  float lmax = -1e30f;
  for (int jj = tid; jj < cnt; jj += 64) {
    const float* kp = Kb + ((size_t)(b * Lk + j0 + jj) * DMODEL) + h * HDIM;
    float s = 0.f;
#pragma unroll 8
    for (int d = 0; d < HDIM; ++d) s += s_q[d] * kp[d];
    s *= 0.125f;  // 1/sqrt(64)
    s_sc[jj] = s;
    lmax = fmaxf(lmax, s);
  }
  s_red[tid] = lmax; __syncthreads();
  for (int o = 32; o >= 1; o >>= 1) {
    if (tid < o) s_red[tid] = fmaxf(s_red[tid], s_red[tid + o]);
    __syncthreads();
  }
  float mx = s_red[0]; __syncthreads();
  float lsum = 0.f;
  for (int jj = tid; jj < cnt; jj += 64) {
    float p = __expf(s_sc[jj] - mx);
    s_sc[jj] = p;
    lsum += p;
  }
  s_red[tid] = lsum; __syncthreads();
  for (int o = 32; o >= 1; o >>= 1) {
    if (tid < o) s_red[tid] += s_red[tid + o];
    __syncthreads();
  }
  float inv = 1.f / s_red[0];
  float acc = 0.f;
  for (int jj = 0; jj < cnt; ++jj)
    acc += s_sc[jj] * Vb[((size_t)(b * Lk + j0 + jj) * DMODEL) + h * HDIM + tid];
  O[((size_t)(b * Lq + q) * DMODEL) + h * HDIM + tid] = acc * inv;
}

// ---------------------------------------------------------------------------
// Host orchestration
// ---------------------------------------------------------------------------

// Param leaf indices (JAX pytree flatten: sorted dict keys, lists in order).
struct SambaIdx {
  int ln1b, ln1s, ln2b, ln2s, ln3b, ln3s, ln4b, ln4s;
  int A, BpB, BpW, CpB, CpW, convB, convW, deltaB, deltaW, inW, outW;
  int m1f1b, m1f1w, m1f2b, m1f2w, m2f1b, m2f1w, m2f2b, m2f2w;
  int skb, skw, sob, sow, sqb, sqw, svb, svw;
};
static SambaIdx mkSamba(int b) {
  SambaIdx s;
  s.ln1b = b + 0;  s.ln1s = b + 1;  s.ln2b = b + 2;  s.ln2s = b + 3;
  s.ln3b = b + 4;  s.ln3s = b + 5;  s.ln4b = b + 6;  s.ln4s = b + 7;
  s.A = b + 8;  s.BpB = b + 9;  s.BpW = b + 10; s.CpB = b + 11; s.CpW = b + 12;
  s.convB = b + 13; s.convW = b + 14; s.deltaB = b + 15; s.deltaW = b + 16;
  s.inW = b + 17; s.outW = b + 18;
  s.m1f1b = b + 19; s.m1f1w = b + 20; s.m1f2b = b + 21; s.m1f2w = b + 22;
  s.m2f1b = b + 23; s.m2f1w = b + 24; s.m2f2b = b + 25; s.m2f2w = b + 26;
  s.skb = b + 27; s.skw = b + 28; s.sob = b + 29; s.sow = b + 30;
  s.sqb = b + 31; s.sqw = b + 32; s.svb = b + 33; s.svw = b + 34;
  return s;
}

struct WS {
  float *X, *ENC, *LN, *XZ, *XC, *IM2, *DEL, *BM, *CM, *H;
  __bf16 *WT;
};

static void gemm(hipStream_t st, const float* A, int M, int K, const float* W, int N,
                 const float* bias, int act, const float* res, float* C, __bf16* wt) {
  int Npad = (N + 63) & ~63;        // pad to 64 for the 16x64 wave tile
  long tot = (long)Npad * K;
  wconv_kernel<<<(unsigned)((tot + 255) / 256), 256, 0, st>>>(W, wt, K, N, Npad);
  int tiles = (M >> 4) * (Npad >> 6);
  gemm_bf16_kernel<<<(tiles + 7) / 8, 256, 0, st>>>(A, wt, bias, res, C, M, K, N, Npad, act);
}

static void run_samba(hipStream_t st, void* const* d_in, const SambaIdx& s,
                      float* x, int Bb, int L, const WS& w) {
  auto P = [&](int i) { return (const float*)d_in[i]; };
  int M = Bb * L;
  // ---- mamba ----
  ln_kernel<<<(M + 7) / 8, 256, 0, st>>>(x, P(s.ln1s), P(s.ln1b), w.LN, M);
  gemm(st, w.LN, M, 512, P(s.inW), 2048, nullptr, 0, nullptr, w.XZ, w.WT);
  {
    long tot = (long)M * 4096;
    im2col_kernel<<<(unsigned)((tot + 255) / 256), 256, 0, st>>>(w.XZ, w.IM2, Bb, L);
  }
  gemm(st, w.IM2, M, 4096, P(s.convW), 1024, P(s.convB), 1, nullptr, w.XC, w.WT);
  gemm(st, w.XC, M, 1024, P(s.deltaW), 1024, P(s.deltaB), 2, nullptr, w.DEL, w.WT);
  gemm(st, w.XC, M, 1024, P(s.BpW), 16, P(s.BpB), 0, nullptr, w.BM, w.WT);
  gemm(st, w.XC, M, 1024, P(s.CpW), 16, P(s.CpB), 0, nullptr, w.CM, w.WT);
  scan_kernel<<<(Bb * DINNER + 255) / 256, 256, 0, st>>>(w.DEL, w.XC, w.BM, w.CM, w.H, Bb, L);
  {
    long tot = (long)M * 1024;
    gate_kernel<<<(unsigned)((tot + 255) / 256), 256, 0, st>>>(w.H, w.XZ, tot);
  }
  gemm(st, w.H, M, 1024, P(s.outW), 512, nullptr, 0, x, x, w.WT);
  // ---- mlp1 ----
  ln_kernel<<<(M + 7) / 8, 256, 0, st>>>(x, P(s.ln2s), P(s.ln2b), w.LN, M);
  gemm(st, w.LN, M, 512, P(s.m1f1w), 2048, P(s.m1f1b), 1, nullptr, w.H, w.WT);
  gemm(st, w.H, M, 2048, P(s.m1f2w), 512, P(s.m1f2b), 0, x, x, w.WT);
  // ---- sliding-window attention ----
  ln_kernel<<<(M + 7) / 8, 256, 0, st>>>(x, P(s.ln3s), P(s.ln3b), w.LN, M);
  float* Qb = w.H;
  float* Kb = w.H + ((size_t)8 << 20) / 4;
  float* Vb = w.H + ((size_t)16 << 20) / 4;
  float* Cx = w.H + ((size_t)24 << 20) / 4;
  gemm(st, w.LN, M, 512, P(s.sqw), 512, P(s.sqb), 0, nullptr, Qb, w.WT);
  gemm(st, w.LN, M, 512, P(s.skw), 512, P(s.skb), 0, nullptr, Kb, w.WT);
  gemm(st, w.LN, M, 512, P(s.svw), 512, P(s.svb), 0, nullptr, Vb, w.WT);
  attn_kernel<<<dim3(L, NHEADS, Bb), 64, 0, st>>>(Qb, Kb, Vb, Cx, L, L, WINDOW_);
  gemm(st, Cx, M, 512, P(s.sow), 512, P(s.sob), 0, x, x, w.WT);
  // ---- mlp2 ----
  ln_kernel<<<(M + 7) / 8, 256, 0, st>>>(x, P(s.ln4s), P(s.ln4b), w.LN, M);
  gemm(st, w.LN, M, 512, P(s.m2f1w), 2048, P(s.m2f1b), 1, nullptr, w.H, w.WT);
  gemm(st, w.H, M, 2048, P(s.m2f2w), 512, P(s.m2f2b), 0, x, x, w.WT);
}

extern "C" void kernel_launch(void* const* d_in, const int* in_sizes, int n_in,
                              void* d_out, int out_size, void* d_ws, size_t ws_size,
                              hipStream_t stream) {
  (void)in_sizes; (void)n_in; (void)out_size; (void)ws_size;
  auto P = [&](int i) { return (const float*)d_in[i]; };
  const int* src_ids = (const int*)d_in[0];
  const int* tgt_ids = (const int*)d_in[1];

  // Leaf index map (base 2): dec[0]=2..46, dec[1]=47..91, dec_ln=92/93,
  // embed=94, enc[0]=95..129, enc[1]=130..164, enc_ln=165/166, lm_head=167/168.
  char* ws = (char*)d_ws;
  size_t off = 0;
  auto carve = [&](size_t bytes) -> char* {
    char* p = ws + off;
    off += (bytes + 255) & ~(size_t)255;
    return p;
  };
  WS w;
  w.WT  = (__bf16*)carve((size_t)30528 * 512 * 2);   // bf16 weight scratch (max: lm_head)
  w.X   = (float*)carve((size_t)4096 * 512 * 4);     // residual stream
  w.ENC = (float*)carve((size_t)4096 * 512 * 4);     // encoder output (normed)
  w.LN  = (float*)carve((size_t)4096 * 512 * 4);
  w.XZ  = (float*)carve((size_t)4096 * 2048 * 4);
  w.XC  = (float*)carve((size_t)4096 * 1024 * 4);
  w.IM2 = (float*)carve((size_t)4096 * 4096 * 4);
  w.DEL = (float*)carve((size_t)4096 * 1024 * 4);
  w.BM  = (float*)carve((size_t)4096 * 16 * 4);
  w.CM  = (float*)carve((size_t)4096 * 16 * 4);
  w.H   = (float*)carve((size_t)4096 * 2048 * 4);    // 32 MB shared scratch

  // -------- encoder --------
  embed_kernel<<<(4096 * 512) / 256, 256, 0, stream>>>(P(94), src_ids, w.X, 4096);
  run_samba(stream, d_in, mkSamba(95), w.X, 2, 2048, w);
  run_samba(stream, d_in, mkSamba(130), w.X, 2, 2048, w);
  ln_kernel<<<4096 / 8, 256, 0, stream>>>(w.X, P(166), P(165), w.ENC, 4096);

  // -------- decoder --------
  embed_kernel<<<(1024 * 512) / 256, 256, 0, stream>>>(P(94), tgt_ids, w.X, 1024);
  int decBase[2] = {2, 47};
  for (int li = 0; li < 2; ++li) {
    int b0 = decBase[li];
    run_samba(stream, d_in, mkSamba(b0 + 8), w.X, 2, 512, w);
    // cross attention
    ln_kernel<<<1024 / 8, 256, 0, stream>>>(w.X, P(b0 + 1), P(b0 + 0), w.LN, 1024);
    float* Qb = w.H;
    float* Kb = w.H + ((size_t)8 << 20) / 4;
    float* Vb = w.H + ((size_t)16 << 20) / 4;
    float* Cx = w.H + ((size_t)24 << 20) / 4;
    gemm(stream, w.LN, 1024, 512, P(b0 + 7), 512, P(b0 + 6), 0, nullptr, Qb, w.WT);   // q
    gemm(stream, w.ENC, 4096, 512, P(b0 + 3), 512, P(b0 + 2), 0, nullptr, Kb, w.WT);  // k
    gemm(stream, w.ENC, 4096, 512, P(b0 + 44), 512, P(b0 + 43), 0, nullptr, Vb, w.WT);// v
    attn_kernel<<<dim3(512, NHEADS, 2), 64, 0, stream>>>(Qb, Kb, Vb, Cx, 512, 2048, -1);
    gemm(stream, Cx, 1024, 512, P(b0 + 5), 512, P(b0 + 4), 0, w.X, w.X, w.WT);        // o + res
  }
  ln_kernel<<<1024 / 8, 256, 0, stream>>>(w.X, P(93), P(92), w.LN, 1024);
  gemm(stream, w.LN, 1024, 512, P(168), VOCAB, P(167), 0, nullptr, (float*)d_out, w.WT);
}
